// CrossFormer_48868137894033
// MI455X (gfx1250) — matrix operand order repaired and checked by reading, not compiled, not run on hardware
//
#include <hip/hip_runtime.h>
#include <math.h>

// ---------------------------------------------------------------------------
// CrossFormer-S forward for gfx1250 (MI455X).
// Heavy GEMMs (qkv / proj / FF / head) -> v_wmma_f32_16x16x32_f16, f32 accum.
// 64x64 block tile, 4 waves, 4 WMMA per wave per K-step, software-pipelined
// branchless staging (float4 global loads + v4h LDS stores on both operands).
// Everything else -> fp32 VALU kernels.
// ---------------------------------------------------------------------------

typedef __attribute__((ext_vector_type(16))) _Float16 v16h;
typedef __attribute__((ext_vector_type(8)))  _Float16 v8h;
typedef __attribute__((ext_vector_type(4)))  _Float16 v4h;
typedef __attribute__((ext_vector_type(8)))  float    v8f;

// ------------------------------ model shape --------------------------------
static const int DIMS_[5]   = {3, 64, 128, 256, 512};
static const int DEPTHS_[4] = {2, 2, 8, 2};
static const int GW_[4]     = {8, 4, 2, 1};     // long (global) window
static const int NK_[4]     = {4, 2, 2, 2};     // # CEL convs per stage
static const int KS_[4][4]  = {{4,8,16,32},{2,4,0,0},{2,4,0,0},{2,4,0,0}};
static const int STR_[4]    = {4, 2, 2, 2};
static const int SP_[4]     = {56, 28, 14, 7};  // spatial size per stage

// ---------------------------------------------------------------------------
// GEMM:  Y[M,N] = W[M,K] * X[K,N] (+bias[M]), fp32 in/out, f16 WMMA compute.
// Batched over blockIdx.z with strides Xbs / Ybs (elements).
// Block: 128 threads = 4 waves; tile 64(M) x 64(N); K stepped by 32.
// Wave w owns M-strip [tileM+16w, +16), computes 4 N-subtiles (4 wmma/step).
// NOTE: X rows may be over-read by <64 floats at the N edge; all X operands
// live in the workspace arena, so those reads stay mapped and are select-zeroed.
// ---------------------------------------------------------------------------
__global__ __launch_bounds__(128)
void gemm_wmma_kernel(const float* __restrict__ Wm, const float* __restrict__ Xm,
                      const float* __restrict__ bias, float* __restrict__ Ym,
                      int M, int N, int K, int Xbs, int Ybs)
{
    __shared__ __align__(16) _Float16 As[64 * 32];   // [m][k]
    __shared__ __align__(16) _Float16 Bs[64 * 32];   // transposed: [n][k]

    const int tid   = threadIdx.x;
    const int lane  = tid & 31;
    const int wave  = tid >> 5;
    const int hi    = lane >> 4;        // half-wave: 0 or 1
    const int lm    = lane & 15;
    const int tileM = blockIdx.y * 64;
    const int tileN = blockIdx.x * 64;

    const float* Xb = Xm + (size_t)blockIdx.z * (size_t)Xbs;
    float*       Yb = Ym + (size_t)blockIdx.z * (size_t)Ybs;

    // staging coordinates
    const int am  = tid >> 3;           // A row within tile (0..15, +16 per j)
    const int ak  = (tid & 7) * 4;      // A k-offset, float4 aligned
    const int bk4 = (tid >> 4) * 4;     // B k-base (0..28), 4 consecutive k
    const int bn4 = (tid & 15) * 4;     // B n-base (0..60), 4 consecutive n

    const bool nAligned = ((N & 3) == 0);
    bool nOk[4];
    #pragma unroll
    for (int e = 0; e < 4; ++e) nOk[e] = (tileN + bn4 + e) < N;

    float aReg[4][4];
    float bReg[4][4];                   // [k-offset][n-offset]

    auto loadA = [&](int k0) {
        #pragma unroll
        for (int j = 0; j < 4; ++j) {
            int  gm  = tileM + am + j * 16;
            bool ok  = gm < M;
            int  gmc = ok ? gm : (M - 1);
            const float4 v = *(const float4*)&Wm[(size_t)gmc * K + (k0 + ak)];
            aReg[j][0] = ok ? v.x : 0.0f;
            aReg[j][1] = ok ? v.y : 0.0f;
            aReg[j][2] = ok ? v.z : 0.0f;
            aReg[j][3] = ok ? v.w : 0.0f;
        }
    };
    auto loadB = [&](int k0) {
        if (nAligned) {                 // vectorized: coalesced float4 over n
            #pragma unroll
            for (int i = 0; i < 4; ++i) {
                const float4 r =
                    *(const float4*)&Xb[(size_t)(k0 + bk4 + i) * N + tileN + bn4];
                bReg[i][0] = nOk[0] ? r.x : 0.0f;
                bReg[i][1] = nOk[1] ? r.y : 0.0f;
                bReg[i][2] = nOk[2] ? r.z : 0.0f;
                bReg[i][3] = nOk[3] ? r.w : 0.0f;
            }
        } else {                        // scalar clamped fallback (N % 4 != 0)
            #pragma unroll
            for (int i = 0; i < 4; ++i) {
                const float* row = &Xb[(size_t)(k0 + bk4 + i) * N];
                #pragma unroll
                for (int e = 0; e < 4; ++e) {
                    int  gn  = tileN + bn4 + e;
                    int  gnc = nOk[e] ? gn : (N - 1);
                    float v  = row[gnc];
                    bReg[i][e] = nOk[e] ? v : 0.0f;
                }
            }
        }
    };
    auto stageA = [&]() {
        #pragma unroll
        for (int j = 0; j < 4; ++j) {
            v4h hv;
            hv[0] = (_Float16)aReg[j][0];
            hv[1] = (_Float16)aReg[j][1];
            hv[2] = (_Float16)aReg[j][2];
            hv[3] = (_Float16)aReg[j][3];
            *(v4h*)&As[(am + j * 16) * 32 + ak] = hv;   // 8B aligned
        }
    };
    auto stageB = [&]() {               // 4x4 in-register transpose -> [n][k]
        #pragma unroll
        for (int e = 0; e < 4; ++e) {
            v4h hv;
            hv[0] = (_Float16)bReg[0][e];
            hv[1] = (_Float16)bReg[1][e];
            hv[2] = (_Float16)bReg[2][e];
            hv[3] = (_Float16)bReg[3][e];
            *(v4h*)&Bs[(bn4 + e) * 32 + bk4] = hv;      // 8B aligned
        }
    };

    v8f acc[4] = {};

    loadA(0);
    loadB(0);
    for (int k0 = 0; k0 < K; k0 += 32) {
        stageA();
        stageB();
        __syncthreads();
        if (k0 + 32 < K) { loadA(k0 + 32); loadB(k0 + 32); }   // pipeline next step

        // ---- A fragment (ISA 7.12.2 16-bit A layout) ----
        // lanes 0-15: e<8 -> K=e,   e>=8 -> K=16+(e-8)
        // lanes16-31: e<8 -> K=8+e, e>=8 -> K=24+(e-8)
        const int koff  = hi * 8;
        const int abase = (wave * 16 + lm) * 32;
        v8h a0 = *(const v8h*)&As[abase + koff];
        v8h a1 = *(const v8h*)&As[abase + 16 + koff];
        v16h af = __builtin_shufflevector(a0, a1, 0,1,2,3,4,5,6,7,8,9,10,11,12,13,14,15);

        #pragma unroll
        for (int t = 0; t < 4; ++t) {
            // B fragment: e<8 -> K=8*hi+e, e>=8 -> K=16+8*hi+(e-8) (matches A)
            const int nb = (t * 16 + lm) * 32 + hi * 8;
            v8h b0 = *(const v8h*)&Bs[nb];
            v8h b1 = *(const v8h*)&Bs[nb + 16];
            v16h bf = __builtin_shufflevector(b0, b1, 0,1,2,3,4,5,6,7,8,9,10,11,12,13,14,15);
            acc[t] = __builtin_amdgcn_wmma_f32_16x16x32_f16(false, af, false, bf,
                                                            (short)0, acc[t], false, false);
        }
        __syncthreads();
    }

    // ---- store C/D: c[r] = C[r + 8*hi][lane&15] ----
    #pragma unroll
    for (int t = 0; t < 4; ++t) {
        int gn = tileN + t * 16 + lm;
        #pragma unroll
        for (int r = 0; r < 8; ++r) {
            int gm = tileM + wave * 16 + r + hi * 8;
            if (gm < M && gn < N) {
                float v = acc[t][r];
                if (bias) v += bias[gm];
                Yb[(size_t)gm * N + gn] = v;
            }
        }
    }
}

// ---------------------------------------------------------------------------
// CEL: direct conv, stride/pad per kernel size, writes its channel slice.
// ---------------------------------------------------------------------------
__global__ void cel_conv_kernel(const float* __restrict__ x, const float* __restrict__ w,
                                const float* __restrict__ bias, float* __restrict__ y,
                                int B, int Cin, int Hin, int Win, int Ctot, int coff,
                                int Cout, int k, int stride, int pad, int Hout, int Wout)
{
    int idx = blockIdx.x * blockDim.x + threadIdx.x;
    int total = B * Cout * Hout * Wout;
    if (idx >= total) return;
    int ow = idx % Wout; int t = idx / Wout;
    int oh = t % Hout;   t /= Hout;
    int oc = t % Cout;   int bb = t / Cout;

    float acc = bias[oc];
    const float* xb = x + (size_t)bb * Cin * Hin * Win;
    const float* wb = w + (size_t)oc * Cin * k * k;
    for (int ic = 0; ic < Cin; ++ic)
        for (int kh = 0; kh < k; ++kh) {
            int ih = oh * stride - pad + kh;
            if (ih < 0 || ih >= Hin) continue;
            for (int kw = 0; kw < k; ++kw) {
                int iw = ow * stride - pad + kw;
                if (iw < 0 || iw >= Win) continue;
                acc += xb[((size_t)ic * Hin + ih) * Win + iw] *
                       wb[((size_t)ic * k + kh) * k + kw];
            }
        }
    y[(((size_t)bb * Ctot + coff + oc) * Hout + oh) * Wout + ow] = acc;
}

// ---------------------------------------------------------------------------
// Channel LayerNorm over C of NCHW (biased var), one thread per pixel.
// ---------------------------------------------------------------------------
__global__ void chan_ln_kernel(const float* __restrict__ x, const float* __restrict__ g,
                               const float* __restrict__ b, float* __restrict__ y,
                               int B, int C, int HW)
{
    int n = blockIdx.x * blockDim.x + threadIdx.x;
    if (n >= B * HW) return;
    int bb = n / HW, hw = n % HW;
    const float* xb = x + (size_t)bb * C * HW + hw;
    float s = 0.0f, s2 = 0.0f;
    for (int c = 0; c < C; ++c) { float v = xb[(size_t)c * HW]; s += v; s2 += v * v; }
    float mean = s / C;
    float var  = s2 / C - mean * mean;
    float inv  = rsqrtf(var + 1e-5f);
    float* yb = y + (size_t)bb * C * HW + hw;
    for (int c = 0; c < C; ++c) {
        float v = xb[(size_t)c * HW];
        yb[(size_t)c * HW] = (v - mean) * inv * g[c] + b[c];
    }
}

// ---------------------------------------------------------------------------
// Dynamic-position-bias MLP: one thread per table row r in [(2w+1)^2].
// ---------------------------------------------------------------------------
__device__ void ln_relu(float* h, const float* g, const float* be, int d4)
{
    float s = 0.0f, s2 = 0.0f;
    for (int j = 0; j < d4; ++j) { s += h[j]; s2 += h[j] * h[j]; }
    float mean = s / d4;
    float var  = s2 / d4 - mean * mean;
    float inv  = rsqrtf(var + 1e-5f);
    for (int j = 0; j < d4; ++j) {
        float v = (h[j] - mean) * inv * g[j] + be[j];
        h[j] = fmaxf(v, 0.0f);
    }
}

__global__ void dpb_kernel(const float* w0, const float* b0, const float* g0, const float* be0,
                           const float* w1, const float* b1, const float* g1, const float* be1,
                           const float* w2, const float* b2, const float* g2, const float* be2,
                           const float* fw, const float* fb, float* __restrict__ biasvals,
                           int wsz, int d4)
{
    int n = 2 * wsz + 1;
    int R = n * n;
    int r = blockIdx.x * blockDim.x + threadIdx.x;
    if (r >= R) return;
    float in0 = (float)(r / n - wsz);
    float in1 = (float)(r % n - wsz);

    float h[128], h2[128];
    for (int j = 0; j < d4; ++j) h[j] = w0[j * 2 + 0] * in0 + w0[j * 2 + 1] * in1 + b0[j];
    ln_relu(h, g0, be0, d4);
    for (int j = 0; j < d4; ++j) {
        float a = b1[j];
        for (int i = 0; i < d4; ++i) a += w1[j * d4 + i] * h[i];
        h2[j] = a;
    }
    ln_relu(h2, g1, be1, d4);
    for (int j = 0; j < d4; ++j) {
        float a = b2[j];
        for (int i = 0; i < d4; ++i) a += w2[j * d4 + i] * h2[i];
        h[j] = a;
    }
    ln_relu(h, g2, be2, d4);
    float out = fb[0];
    for (int j = 0; j < d4; ++j) out += fw[j] * h[j];
    biasvals[r] = out;
}

// ---------------------------------------------------------------------------
// Windowed attention core. One block per (batch, window, head); thread = query
// token. Window gather: short h=wy*w+ty ; long (dilated) h=ty*nh+wy.
// qkv: (B, 3d, S, S) full-image (pointwise qkv commutes with windowing).
// ---------------------------------------------------------------------------
__global__ void attn_core_kernel(const float* __restrict__ qkv,
                                 const float* __restrict__ biasvals,
                                 float* __restrict__ out,
                                 int d, int S, int wsz, int nh, int heads, int isLong)
{
    __shared__ float Ks[64][32];
    __shared__ float Vs[64][32];

    const int T   = wsz * wsz;
    const int bid = blockIdx.x;
    const int h   = bid % heads;
    const int win = (bid / heads) % (nh * nh);
    const int bb  = bid / (heads * nh * nh);
    const int wy  = win / nh, wx = win % nh;
    const int t   = threadIdx.x;
    const int SS  = S * S;

    float q[32];
    int ph = 0, pw = 0;
    if (t < T) {
        int ty = t / wsz, tx = t % wsz;
        if (isLong) { ph = ty * nh + wy; pw = tx * nh + wx; }
        else        { ph = wy * wsz + ty; pw = wx * wsz + tx; }
        const float* base = qkv + (size_t)bb * 3 * d * SS + (size_t)(ph * S + pw);
        size_t hc = (size_t)h * 32;
        #pragma unroll
        for (int c = 0; c < 32; ++c) {
            q[c]     = base[(hc + c) * SS] * 0.17677669529663687f;   // 32^-0.5
            Ks[t][c] = base[((size_t)d + hc + c) * SS];
            Vs[t][c] = base[((size_t)2 * d + hc + c) * SS];
        }
    }
    __syncthreads();
    if (t >= T) return;

    const int yi = t / wsz, xi = t % wsz;
    const int n2 = 2 * wsz - 1;

    float mx = -1e30f;
    for (int j = 0; j < T; ++j) {
        float s = 0.0f;
        #pragma unroll
        for (int c = 0; c < 32; ++c) s += q[c] * Ks[j][c];
        int yj = j / wsz, xj = j % wsz;
        s += biasvals[(yi - yj + wsz - 1) * n2 + (xi - xj + wsz - 1)];
        mx = fmaxf(mx, s);
    }
    float o[32];
    #pragma unroll
    for (int c = 0; c < 32; ++c) o[c] = 0.0f;
    float Z = 0.0f;
    for (int j = 0; j < T; ++j) {
        float s = 0.0f;
        #pragma unroll
        for (int c = 0; c < 32; ++c) s += q[c] * Ks[j][c];
        int yj = j / wsz, xj = j % wsz;
        s += biasvals[(yi - yj + wsz - 1) * n2 + (xi - xj + wsz - 1)];
        float e = expf(s - mx);
        Z += e;
        #pragma unroll
        for (int c = 0; c < 32; ++c) o[c] += e * Vs[j][c];
    }
    float invZ = 1.0f / Z;
    float* ob = out + (size_t)bb * d * SS + (size_t)(ph * S + pw);
    size_t hc = (size_t)h * 32;
    #pragma unroll
    for (int c = 0; c < 32; ++c) ob[(hc + c) * SS] = o[c] * invZ;
}

// ------------------------------ elementwise --------------------------------
__global__ void add_kernel(float* __restrict__ x, const float* __restrict__ t, int n)
{
    int i = blockIdx.x * blockDim.x + threadIdx.x;
    if (i < n) x[i] += t[i];
}

__global__ void gelu_kernel(float* __restrict__ x, int n)
{
    int i = blockIdx.x * blockDim.x + threadIdx.x;
    if (i < n) {
        float v = x[i];
        x[i] = 0.5f * v * (1.0f + erff(v * 0.70710678118654752f));
    }
}

// pooled written K-major: P[c*B + b] so head GEMM sees X[K,N] with N=batch
__global__ void pool_kernel(const float* __restrict__ x, float* __restrict__ p,
                            int B, int C, int HW)
{
    int i = blockIdx.x * blockDim.x + threadIdx.x;
    if (i >= B * C) return;
    int bb = i / C, c = i % C;
    const float* xb = x + ((size_t)bb * C + c) * HW;
    float s = 0.0f;
    for (int k = 0; k < HW; ++k) s += xb[k];
    p[(size_t)c * B + bb] = s / HW;
}

__global__ void head_out_kernel(const float* __restrict__ HY, float* __restrict__ out)
{
    int i = blockIdx.x * blockDim.x + threadIdx.x;
    if (i >= 8 * 1000) return;
    int b = i / 1000, m = i % 1000;
    out[i] = HY[(size_t)m * 8 + b];
}

// ---------------------------------------------------------------------------
// Host-side parameter index map (pytree flatten order is ambiguous; we match
// in_sizes against insertion-order and jax-sorted-key order, x first or last).
// ---------------------------------------------------------------------------
struct AttnIdx { int ng, nb, qkv, ow, ob, fw, fb, lw[3], lb[3], lg[3], lbe[3]; };
struct FFIdx   { int ng, nb, w1, b1, w2, b2; };
struct BlockIdx{ AttnIdx sa, la; FFIdx sf, lf; };
struct NetIdx  { int x, headw, headb; int celw[4][4], celb[4][4]; BlockIdx blk[4][8]; };

struct Builder {
    const int* sizes; int n; int cnt; bool ok;
    int add(int sz) {
        int i = cnt++;
        if (i >= n || sizes[i] != sz) ok = false;
        return i;
    }
};

static void celSplit(int s, int d, int* dsc)
{
    int L = NK_[s], sum = 0;
    for (int i = 0; i < L - 1; ++i) { dsc[i] = d >> (i + 1); sum += dsc[i]; }
    dsc[L - 1] = d - sum;
}

static void fillAttn(Builder& B, AttnIdx& a, int d, bool sorted)
{
    int d4 = d / 4;
    if (!sorted) {   // insertion: ng, nb, qkv, ow, ob, dpb{layers{w,b,g,be}, fw, fb}
        a.ng = B.add(d); a.nb = B.add(d); a.qkv = B.add(3 * d * d);
        a.ow = B.add(d * d); a.ob = B.add(d);
        for (int l = 0; l < 3; ++l) {
            int fin = l ? d4 : 2;
            a.lw[l] = B.add(d4 * fin); a.lb[l] = B.add(d4);
            a.lg[l] = B.add(d4); a.lbe[l] = B.add(d4);
        }
        a.fw = B.add(d4); a.fb = B.add(1);
    } else {         // sorted: dpb{fb, fw, layers{b,be,g,w}}, nb, ng, ob, ow, qkv
        a.fb = B.add(1); a.fw = B.add(d4);
        for (int l = 0; l < 3; ++l) {
            int fin = l ? d4 : 2;
            a.lb[l] = B.add(d4); a.lbe[l] = B.add(d4);
            a.lg[l] = B.add(d4); a.lw[l] = B.add(d4 * fin);
        }
        a.nb = B.add(d); a.ng = B.add(d); a.ob = B.add(d);
        a.ow = B.add(d * d); a.qkv = B.add(3 * d * d);
    }
}

static void fillFF(Builder& B, FFIdx& f, int d, bool sorted)
{
    if (!sorted) {   // insertion: ng, nb, w1, b1, w2, b2
        f.ng = B.add(d); f.nb = B.add(d);
        f.w1 = B.add(4 * d * d); f.b1 = B.add(4 * d);
        f.w2 = B.add(4 * d * d); f.b2 = B.add(d);
    } else {         // sorted: b1, b2, nb, ng, w1, w2
        f.b1 = B.add(4 * d); f.b2 = B.add(d);
        f.nb = B.add(d); f.ng = B.add(d);
        f.w1 = B.add(4 * d * d); f.w2 = B.add(4 * d * d);
    }
}

// order 0: insertion, x first. order 1: sorted keys, x first. order 2: sorted, x last.
static bool buildNet(NetIdx& net, const int* in_sizes, int n_in, int order)
{
    Builder B{in_sizes, n_in, 0, true};
    const int XSZ = 8 * 3 * 224 * 224;
    if (order != 2) net.x = B.add(XSZ);
    bool sorted = (order != 0);
    if (sorted) { net.headb = B.add(1000); net.headw = B.add(1000 * 512); }
    for (int s = 0; s < 4; ++s) {
        int din = DIMS_[s], d = DIMS_[s + 1];
        int dsc[4]; celSplit(s, d, dsc);
        if (!sorted) {
            for (int i = 0; i < NK_[s]; ++i) {
                int k = KS_[s][i];
                net.celw[s][i] = B.add(dsc[i] * din * k * k);
                net.celb[s][i] = B.add(dsc[i]);
            }
            for (int bi = 0; bi < DEPTHS_[s]; ++bi) {
                fillAttn(B, net.blk[s][bi].sa, d, false);
                fillFF  (B, net.blk[s][bi].sf, d, false);
                fillAttn(B, net.blk[s][bi].la, d, false);
                fillFF  (B, net.blk[s][bi].lf, d, false);
            }
        } else {      // sorted: blocks first (keys: la, lf, sa, sf), then cel {b, w}
            for (int bi = 0; bi < DEPTHS_[s]; ++bi) {
                fillAttn(B, net.blk[s][bi].la, d, true);
                fillFF  (B, net.blk[s][bi].lf, d, true);
                fillAttn(B, net.blk[s][bi].sa, d, true);
                fillFF  (B, net.blk[s][bi].sf, d, true);
            }
            for (int i = 0; i < NK_[s]; ++i) {
                int k = KS_[s][i];
                net.celb[s][i] = B.add(dsc[i]);
                net.celw[s][i] = B.add(dsc[i] * din * k * k);
            }
        }
    }
    if (!sorted) { net.headw = B.add(1000 * 512); net.headb = B.add(1000); }
    if (order == 2) net.x = B.add(XSZ);
    return B.ok && B.cnt == n_in;
}

// ---------------------------------------------------------------------------
extern "C" void kernel_launch(void* const* d_in, const int* in_sizes, int n_in,
                              void* d_out, int out_size, void* d_ws, size_t ws_size,
                              hipStream_t stream)
{
    (void)out_size;
    if (n_in < 3) return;

    NetIdx net;
    bool matched = false;
    for (int o = 0; o < 3 && !matched; ++o) matched = buildNet(net, in_sizes, n_in, o);
    if (!matched) {
        if (n_in < 723) return;               // structure unknown; cannot proceed
        buildNet(net, in_sizes, n_in, 0);     // fall back to insertion order
    }

    auto in = [&](int i) { return (const float*)d_in[i]; };

    // ----- workspace arena (floats) -----
    const size_t SZX  = 1605632;   // 8*64*56*56 (largest activation)
    const size_t SZT2 = 6422528;   // 8*256*56*56 (largest FF hidden / qkv)
    float* ws    = (float*)d_ws;
    float* XA    = ws;
    float* XB    = XA + SZX;
    float* T1    = XB + SZX;       // LN out / proj out / FF out
    float* T3    = T1 + SZX;       // attention-core output
    float* T2    = T3 + SZX;       // qkv / FF hidden
    float* BIASV = T2 + SZT2;      // DPB bias table ((2w+1)^2 <= 289)
    float* P     = BIASV + 512;    // pooled, K-major (512 x 8)
    float* HY    = P + 512 * 8;    // head logits (1000 x 8)
    size_t need  = (size_t)((HY + 8 * 1000 + 64) - ws) * sizeof(float);
    if (ws_size < need) return;

    auto gemm = [&](const float* Wm, const float* Xm, const float* bias, float* Ym,
                    int M, int N, int K, int Xbs, int Ybs, int nb) {
        dim3 g((N + 63) / 64, (M + 63) / 64, nb);
        gemm_wmma_kernel<<<g, dim3(128), 0, stream>>>(Wm, Xm, bias, Ym, M, N, K, Xbs, Ybs);
    };

    auto runAttn = [&](const AttnIdx& A, int wsz, int isLong, int d, int S, float* X) {
        int NN = S * S, heads = d / 32, d4 = d / 4, nh = S / wsz;
        int totpx = 8 * NN;
        chan_ln_kernel<<<(totpx + 255) / 256, 256, 0, stream>>>(
            X, in(A.ng), in(A.nb), T1, 8, d, NN);
        gemm(in(A.qkv), T1, nullptr, T2, 3 * d, NN, d, d * NN, 3 * d * NN, 8);
        dpb_kernel<<<1, 256, 0, stream>>>(
            in(A.lw[0]), in(A.lb[0]), in(A.lg[0]), in(A.lbe[0]),
            in(A.lw[1]), in(A.lb[1]), in(A.lg[1]), in(A.lbe[1]),
            in(A.lw[2]), in(A.lb[2]), in(A.lg[2]), in(A.lbe[2]),
            in(A.fw), in(A.fb), BIASV, wsz, d4);
        int nblocks = 8 * nh * nh * heads;
        attn_core_kernel<<<nblocks, 64, 0, stream>>>(
            T2, BIASV, T3, d, S, wsz, nh, heads, isLong);
        gemm(in(A.ow), T3, in(A.ob), T1, d, NN, d, d * NN, d * NN, 8);
        int tot = 8 * d * NN;
        add_kernel<<<(tot + 255) / 256, 256, 0, stream>>>(X, T1, tot);
    };

    auto runFF = [&](const FFIdx& F, int d, int S, float* X) {
        int NN = S * S, totpx = 8 * NN;
        chan_ln_kernel<<<(totpx + 255) / 256, 256, 0, stream>>>(
            X, in(F.ng), in(F.nb), T1, 8, d, NN);
        gemm(in(F.w1), T1, in(F.b1), T2, 4 * d, NN, d, d * NN, 4 * d * NN, 8);
        int th = 8 * 4 * d * NN;
        gelu_kernel<<<(th + 255) / 256, 256, 0, stream>>>(T2, th);
        gemm(in(F.w2), T2, in(F.b2), T1, d, NN, 4 * d, 4 * d * NN, d * NN, 8);
        int tot = 8 * d * NN;
        add_kernel<<<(tot + 255) / 256, 256, 0, stream>>>(X, T1, tot);
    };

    // ----- stages -----
    const float* src = in(net.x);
    float* X = XA;
    float* Xother = XB;
    int Hin = 224, Win = 224, din = 3;
    for (int s = 0; s < 4; ++s) {
        int d = DIMS_[s + 1], S = SP_[s];
        int dsc[4]; celSplit(s, d, dsc);
        int coff = 0;
        for (int i = 0; i < NK_[s]; ++i) {
            int k = KS_[s][i];
            int pad = (k - STR_[s]) / 2;
            int tot = 8 * dsc[i] * S * S;
            cel_conv_kernel<<<(tot + 255) / 256, 256, 0, stream>>>(
                src, in(net.celw[s][i]), in(net.celb[s][i]), X,
                8, din, Hin, Win, d, coff, dsc[i], k, STR_[s], pad, S, S);
            coff += dsc[i];
        }
        for (int bi = 0; bi < DEPTHS_[s]; ++bi) {
            const BlockIdx& bk = net.blk[s][bi];
            runAttn(bk.sa, 7,      0, d, S, X);   // short / local
            runFF  (bk.sf,            d, S, X);
            runAttn(bk.la, GW_[s], 1, d, S, X);   // long / dilated
            runFF  (bk.lf,            d, S, X);
        }
        src = X;
        float* tmp = Xother; Xother = X; X = tmp;
        Hin = S; Win = S; din = d;
    }

    // ----- head -----
    pool_kernel<<<(8 * 512 + 255) / 256, 256, 0, stream>>>(src, P, 8, 512, 49);
    gemm(in(net.headw), P, in(net.headb), HY, 1000, 8, 512, 0, 0, 1);
    head_out_kernel<<<(8000 + 255) / 256, 256, 0, stream>>>(HY, (float*)d_out);
}
